// SHSAttention_24945170055281
// MI455X (gfx1250) — compile-verified, matrix operand order checked by reading, not run on hardware
//
#include <hip/hip_runtime.h>

// ---------------------------------------------------------------------------
// Single-head attention for MI455X (gfx1250), wave32 + WMMA bf16.
//   Stage 1: Q = (X@Wq)*scale, K = X@Wk  (bf16 row-major [S][D])
//            Vt = (X@Wv)^T              (bf16 [D][S])
//   Stage 2: flash attention, one wave per 16 query rows, key blocks of 32.
// ---------------------------------------------------------------------------

typedef __attribute__((ext_vector_type(16))) __bf16 v16bf;
typedef __attribute__((ext_vector_type(16))) float  v16f;
typedef __attribute__((ext_vector_type(8)))  float  v8f;

static constexpr int kSEQ = 8192;
static constexpr int kEMB = 1024;
static constexpr int kD   = 128;
static constexpr int PSTR = 40;   // padded LDS row stride (bf16 elems) -> conflict-free b128 reads

union Frag16 {
    v16bf v;
    uint4 q[2];
    unsigned short u16[16];
};

union FragF {
    v16f   v;
    float4 q[4];
    float  f[16];
};

static __device__ __forceinline__ unsigned short bf16bits(float f) {
    return __builtin_bit_cast(unsigned short, (__bf16)f);
}

static __device__ __forceinline__ v8f wmma_bf16(const v16bf& a, const v16bf& b, const v8f& c) {
    return __builtin_amdgcn_wmma_f32_16x16x32_bf16(
        /*neg_a=*/false, a, /*neg_b=*/false, b,
        /*c_mod=*/(short)0, c, /*reuse_a=*/false, /*reuse_b=*/false);
}

// ---------------------------------------------------------------------------
// Stage 1: projection GEMM. Block = 256 threads = 8 waves; wave w owns the
// 16-column N-tile w; block owns 16 rows. W k-chunks are staged to LDS
// transposed (each thread owns one column -> contiguous b128 DS stores,
// packed f32->bf16 conversions) so B fragments are contiguous 16B ds loads.
// mode 0/1: out[row][col] (bf16, row-major), mode 2: out[col][row] (Vt).
// ---------------------------------------------------------------------------
__global__ __launch_bounds__(256) void proj_kernel(const float* __restrict__ X,
                                                   const float* __restrict__ W,
                                                   unsigned short* __restrict__ out,
                                                   int mode, float scale) {
    __shared__ unsigned short Wt[kD * PSTR];   // [col][k_local], k_local in [0,32)

    const int lane  = threadIdx.x & 31;
    const int wv    = threadIdx.x >> 5;        // N tile 0..7
    const int m0    = blockIdx.x * 16;
    const int nl    = lane & 15;
    const int hi    = lane >> 4;
    const int kbase = hi * 8;                  // lane-half k offset (A/B layout)

    v8f acc = {};

    for (int kb = 0; kb < kEMB; kb += 32) {
        __syncthreads();
        {   // cooperative staging: 32 k-rows x 128 cols of W -> LDS transposed bf16.
            // Thread owns one column and a 16-row k-half: strided global b32 loads
            // (coalesced across the 128 column-adjacent threads), packed cvt,
            // two contiguous ds_store_b128.
            const int c  = threadIdx.x & 127;   // column 0..127
            const int h  = threadIdx.x >> 7;    // k half 0..1
            const int k0 = h * 16;
            FragF wf;
#pragma unroll
            for (int i = 0; i < 16; ++i)
                wf.f[i] = W[(kb + k0 + i) * kD + c];
            Frag16 wb;
            wb.v = __builtin_convertvector(wf.v, v16bf);
            unsigned short* dst = Wt + c * PSTR + k0;
            *(uint4*)(dst)     = wb.q[0];
            *(uint4*)(dst + 8) = wb.q[1];
        }
        __syncthreads();

        // A fragment: X rows m0..m0+15, cols kb..kb+31 (f32 -> bf16, packed cvt)
        FragF xf;
        const float* xp = X + (m0 + nl) * kEMB + kb + kbase;
        xf.q[0] = *(const float4*)(xp + 0);
        xf.q[1] = *(const float4*)(xp + 4);
        xf.q[2] = *(const float4*)(xp + 16);
        xf.q[3] = *(const float4*)(xp + 20);
        Frag16 fa;
        fa.v = __builtin_convertvector(xf.v, v16bf);

        // B fragment from transposed LDS: contiguous 16B chunks
        Frag16 fb;
        const unsigned short* wp = Wt + (wv * 16 + nl) * PSTR + kbase;
        fb.q[0] = *(const uint4*)(wp);
        fb.q[1] = *(const uint4*)(wp + 16);

        acc = wmma_bf16(fa.v, fb.v, acc);
    }

#pragma unroll
    for (int r = 0; r < 8; ++r) {
        const int row = m0 + r + 8 * hi;
        const int col = wv * 16 + nl;
        const unsigned short h = bf16bits(acc[r] * scale);
        if (mode == 2) out[col * kSEQ + row] = h;   // Vt
        else           out[row * kD + col]  = h;    // Q (pre-scaled) / K
    }
}

// ---------------------------------------------------------------------------
// Stage 2: flash attention. One wave = 16 query rows x full D=128.
// Key block = 32  ->  P is exactly one 16x32 bf16 A fragment.
// ---------------------------------------------------------------------------
__global__ __launch_bounds__(256) void attn_kernel(const unsigned short* __restrict__ Qb,
                                                   const unsigned short* __restrict__ Kb,
                                                   const unsigned short* __restrict__ Vt,
                                                   float* __restrict__ Out) {
    __shared__ unsigned short Plds[8 * 16 * PSTR];   // per-wave 16x32 P tile (padded)

    const int lane  = threadIdx.x & 31;
    const int wv    = threadIdx.x >> 5;
    const int m0    = (blockIdx.x * 8 + wv) * 16;
    const int nl    = lane & 15;
    const int hi    = lane >> 4;
    const int kbase = hi * 8;
    unsigned short* Pw = Plds + wv * 16 * PSTR;

    // Q A-fragments (bf16, pre-scaled by 1/sqrt(D)): 4 x (16x32)
    Frag16 aQ[4];
#pragma unroll
    for (int t = 0; t < 4; ++t) {
        const unsigned short* qp = Qb + (m0 + nl) * kD + t * 32 + kbase;
        aQ[t].q[0] = *(const uint4*)(qp);
        aQ[t].q[1] = *(const uint4*)(qp + 16);
    }

    v8f accO[8];
#pragma unroll
    for (int n = 0; n < 8; ++n) accO[n] = (v8f){};
    float m_i[8], l_i[8];
#pragma unroll
    for (int r = 0; r < 8; ++r) { m_i[r] = -1.0e30f; l_i[r] = 0.0f; }

    for (int j = 0; j < kSEQ; j += 32) {
        // ---- S = Q @ K^T for 32 keys: two 16x16 column tiles, 4 k-steps each
        v8f S0 = (v8f){}, S1 = (v8f){};
#pragma unroll
        for (int t = 0; t < 4; ++t) {
            Frag16 b0, b1;
            const unsigned short* kp0 = Kb + (j + nl) * kD + t * 32 + kbase;
            b0.q[0] = *(const uint4*)(kp0);
            b0.q[1] = *(const uint4*)(kp0 + 16);
            const unsigned short* kp1 = Kb + (j + 16 + nl) * kD + t * 32 + kbase;
            b1.q[0] = *(const uint4*)(kp1);
            b1.q[1] = *(const uint4*)(kp1 + 16);
            S0 = wmma_bf16(aQ[t].v, b0.v, S0);
            S1 = wmma_bf16(aQ[t].v, b1.v, S1);
        }

        if (j + 32 < kSEQ) {
            __builtin_prefetch(Kb + (j + 32 + nl) * kD, 0, 1);
            __builtin_prefetch(Vt + nl * kSEQ + j + 32, 0, 1);
        }

        // ---- online softmax (row = one 16-lane half, per accumulator elem)
        float p0[8], p1[8];
#pragma unroll
        for (int r = 0; r < 8; ++r) {
            float rm = fmaxf(S0[r], S1[r]);
#pragma unroll
            for (int mask = 1; mask < 16; mask <<= 1)
                rm = fmaxf(rm, __shfl_xor(rm, mask));
            const float mn = fmaxf(m_i[r], rm);
            const float al = __expf(m_i[r] - mn);
            p0[r] = __expf(S0[r] - mn);
            p1[r] = __expf(S1[r] - mn);
            float rs = p0[r] + p1[r];
#pragma unroll
            for (int mask = 1; mask < 16; mask <<= 1)
                rs += __shfl_xor(rs, mask);
            l_i[r] = l_i[r] * al + rs;
            m_i[r] = mn;
#pragma unroll
            for (int n = 0; n < 8; ++n) accO[n][r] *= al;
        }

        // ---- C-layout -> A-layout transpose of P through per-wave LDS
#pragma unroll
        for (int r = 0; r < 8; ++r) {
            const int row = r + 8 * hi;
            Pw[row * PSTR + nl]      = bf16bits(p0[r]);
            Pw[row * PSTR + 16 + nl] = bf16bits(p1[r]);
        }
        asm volatile("s_wait_dscnt 0" ::: "memory");   // wave-private: DS order only

        Frag16 aP;
        const unsigned short* pp = Pw + nl * PSTR + kbase;
        aP.q[0] = *(const uint4*)(pp);
        aP.q[1] = *(const uint4*)(pp + 16);

        // ---- O += P @ V  (V transposed in memory -> contiguous B fragments)
#pragma unroll
        for (int n = 0; n < 8; ++n) {
            Frag16 bV;
            const unsigned short* vp = Vt + (n * 16 + nl) * kSEQ + j + kbase;
            bV.q[0] = *(const uint4*)(vp);
            bV.q[1] = *(const uint4*)(vp + 16);
            accO[n] = wmma_bf16(aP.v, bV.v, accO[n]);
        }
    }

    // ---- normalize and store f32 output (one reciprocal per row)
#pragma unroll
    for (int r = 0; r < 8; ++r) {
        const float inv = 1.0f / l_i[r];
        const int   row = (m0 + r + 8 * hi) * kD;
#pragma unroll
        for (int n = 0; n < 8; ++n)
            Out[row + n * 16 + nl] = accO[n][r] * inv;
    }
}

// ---------------------------------------------------------------------------
extern "C" void kernel_launch(void* const* d_in, const int* in_sizes, int n_in,
                              void* d_out, int out_size, void* d_ws, size_t ws_size,
                              hipStream_t stream) {
    (void)in_sizes; (void)n_in; (void)out_size; (void)ws_size;
    const float* x   = (const float*)d_in[0];
    const float* wq  = (const float*)d_in[1];
    const float* wk  = (const float*)d_in[2];
    const float* wvv = (const float*)d_in[3];

    unsigned short* Qb = (unsigned short*)d_ws;          // [SEQ][D] bf16
    unsigned short* Kb = Qb + (size_t)kSEQ * kD;         // [SEQ][D] bf16
    unsigned short* Vt = Kb + (size_t)kSEQ * kD;         // [D][SEQ] bf16
    float* out = (float*)d_out;

    const float scale = 0.08838834764831845f;            // 1/sqrt(128)
    dim3 blk(256);

    proj_kernel<<<kSEQ / 16, blk, 0, stream>>>(x, wq, Qb, 0, scale);
    proj_kernel<<<kSEQ / 16, blk, 0, stream>>>(x, wk, Kb, 1, 1.0f);
    proj_kernel<<<kSEQ / 16, blk, 0, stream>>>(x, wvv, Vt, 2, 1.0f);
    attn_kernel<<<kSEQ / (16 * 8), blk, 0, stream>>>(Qb, Kb, Vt, out);
}
